// VectorQuantizer_38671885533486
// MI455X (gfx1250) — matrix operand order repaired and checked by reading, not compiled, not run on hardware
//
#include <hip/hip_runtime.h>

typedef _Float16 v16h __attribute__((ext_vector_type(16)));
typedef _Float16 v8h  __attribute__((ext_vector_type(8)));
typedef float    v8f  __attribute__((ext_vector_type(8)));
typedef int      v4i  __attribute__((ext_vector_type(4)));

#define FLT_MAX_C 3.402823466e+38f

#define D_DIM    128
#define K_CODES  1024
#define N_TOK    131072            // B*T = 32*4096
#define KCHUNK   64                // codes staged per chunk (fits raw+f16 in <64KB LDS)
#define NCHUNK   (K_CODES / KCHUNK)
#define ROW_PAD  136               // halves per staged f16 row: 128 + 8 pad (4-bank shift/row)
#define NBLOCKS  (N_TOK / 256)     // 512 blocks, 256 rows each

typedef __attribute__((address_space(1))) v4i g_v4i;   // global 128-bit payload
typedef __attribute__((address_space(3))) v4i l_v4i;   // LDS 128-bit payload

// ---- CDNA5 async global->LDS copy (ASYNCcnt-tracked), builtin or asm fallback ----
__device__ __forceinline__ void async_g2l_b128(const void* gsrc, void* ldst) {
#if __has_builtin(__builtin_amdgcn_global_load_async_to_lds_b128)
  __builtin_amdgcn_global_load_async_to_lds_b128(
      (g_v4i*)(void*)gsrc, (l_v4i*)ldst, 0, 0);
#else
  unsigned laddr = (unsigned)(unsigned long long)(l_v4i*)ldst;
  asm volatile("global_load_async_to_lds_b128 %0, %1, off"
               :: "v"(laddr), "v"(gsrc) : "memory");
#endif
}

__device__ __forceinline__ void wait_async0() {
#if __has_builtin(__builtin_amdgcn_s_wait_asynccnt)
  __builtin_amdgcn_s_wait_asynccnt(0);
#else
  asm volatile("s_wait_asynccnt 0" ::: "memory");
#endif
}

__device__ __forceinline__ v8f wmma16x16x32(v16h a, v16h b, v8f c) {
  // D = A(16x32 f16) * B(32x16 f16) + C(16x16 f32)
  return __builtin_amdgcn_wmma_f32_16x16x32_f16(false, a, false, b, (short)0, c,
                                                false, false);
}

__global__ __launch_bounds__(256) void vq_main_kernel(
    const float* __restrict__ x, const float* __restrict__ emb,
    float* __restrict__ q_out, float* __restrict__ idx_out,
    float* __restrict__ part) {
  __shared__ __align__(16) float    ldsRaw[KCHUNK * D_DIM];    // 32 KB async landing pad
  __shared__ __align__(16) _Float16 ldsE[KCHUNK * ROW_PAD];    // 17 KB f16 fragments
  __shared__ float ldsNormPart[256];
  __shared__ float ldsENorm[KCHUNK];
  __shared__ int   ldsIdx[256];
  __shared__ float ldsRed[8];

  const int t    = threadIdx.x;
  const int lane = t & 31;
  const int wave = t >> 5;
  const int lmod = lane & 15;
  const int hh   = lane >> 4;                 // 0: lanes 0-15, 1: lanes 16-31
  const int rowBase = blockIdx.x * 256;
  const int mBase   = rowBase + wave * 32;

  // ---- load + convert this wave's 32 input rows into WMMA A fragments ----
  // 16-bit A 16x32 layout: half j -> k = (j&8)*2 + 8*hh + (j&7)
  v16h a[2][4];
#pragma unroll
  for (int mt = 0; mt < 2; ++mt) {
    const float* xr = x + (size_t)(mBase + mt * 16 + lmod) * D_DIM;
#pragma unroll
    for (int c = 0; c < 4; ++c) {
      const int dbase = c * 32 + hh * 8;
      float4 f0 = *(const float4*)(xr + dbase + 0);
      float4 f1 = *(const float4*)(xr + dbase + 4);
      float4 f2 = *(const float4*)(xr + dbase + 16);
      float4 f3 = *(const float4*)(xr + dbase + 20);
      v16h af;
      af[0]  = (_Float16)f0.x; af[1]  = (_Float16)f0.y;
      af[2]  = (_Float16)f0.z; af[3]  = (_Float16)f0.w;
      af[4]  = (_Float16)f1.x; af[5]  = (_Float16)f1.y;
      af[6]  = (_Float16)f1.z; af[7]  = (_Float16)f1.w;
      af[8]  = (_Float16)f2.x; af[9]  = (_Float16)f2.y;
      af[10] = (_Float16)f2.z; af[11] = (_Float16)f2.w;
      af[12] = (_Float16)f3.x; af[13] = (_Float16)f3.y;
      af[14] = (_Float16)f3.z; af[15] = (_Float16)f3.w;
      a[mt][c] = af;
    }
  }

  float vmin[2][8];
  int   vidx[2][8];
#pragma unroll
  for (int mt = 0; mt < 2; ++mt)
#pragma unroll
    for (int r = 0; r < 8; ++r) { vmin[mt][r] = FLT_MAX_C; vidx[mt][r] = 0; }

  // ---- software-pipelined codebook sweep: async global->LDS overlaps WMMA ----
  // prologue: launch chunk 0 (8 b128 transfers per thread = 32 KB per block)
#pragma unroll
  for (int i = 0; i < 8; ++i) {
    const int f4 = i * 256 + t;
    async_g2l_b128(emb + f4 * 4, ldsRaw + f4 * 4);
  }

  for (int ch = 0; ch < NCHUNK; ++ch) {
    wait_async0();        // this wave's async transfers landed
    __syncthreads();      // all waves' transfers landed; prev compute done

    // convert staged f32 rows -> padded f16 LDS + exact f32 row norms
    {
      const int r = t >> 2, h = t & 3;           // 4 threads per code row
      const float* src = ldsRaw + r * D_DIM + h * 32;
      _Float16* dst = ldsE + r * ROW_PAD + h * 32;
      float s = 0.f;
#pragma unroll
      for (int i = 0; i < 32; i += 4) {
        float4 v = *(const float4*)(src + i);
        s += v.x * v.x + v.y * v.y + v.z * v.z + v.w * v.w;
        dst[i + 0] = (_Float16)v.x; dst[i + 1] = (_Float16)v.y;
        dst[i + 2] = (_Float16)v.z; dst[i + 3] = (_Float16)v.w;
      }
      ldsNormPart[t] = s;
    }
    __syncthreads();
    if (t < KCHUNK)
      ldsENorm[t] = (ldsNormPart[4 * t] + ldsNormPart[4 * t + 1]) +
                    (ldsNormPart[4 * t + 2] + ldsNormPart[4 * t + 3]);
    __syncthreads();      // conversion reads of ldsRaw complete -> safe to refill

    // overlap: launch next chunk's global->LDS while we run the WMMA loop
    if (ch + 1 < NCHUNK) {
      const float* gnext = emb + (size_t)(ch + 1) * KCHUNK * D_DIM;
#pragma unroll
      for (int i = 0; i < 8; ++i) {
        const int f4 = i * 256 + t;
        async_g2l_b128(gnext + f4 * 4, ldsRaw + f4 * 4);
      }
    }

    // ---- score 16-code column tiles: 8 WMMAs per tile (2 M-tiles x 4 K-steps) ----
#pragma unroll
    for (int ct = 0; ct < KCHUNK / 16; ++ct) {
      // 16-bit B 32x16 layout: half j -> k = (j&8)*2 + 8*hh + (j&7), n = lmod
      const _Float16* brow = ldsE + (ct * 16 + lmod) * ROW_PAD + hh * 8;
      v8f acc0 = {0.f, 0.f, 0.f, 0.f, 0.f, 0.f, 0.f, 0.f};
      v8f acc1 = acc0;
#pragma unroll
      for (int c = 0; c < 4; ++c) {
        v8h b0 = *(const v8h*)(brow + c * 32);
        v8h b1 = *(const v8h*)(brow + c * 32 + 16);
        v16h bf = __builtin_shufflevector(b0, b1, 0, 1, 2, 3, 4, 5, 6, 7,
                                          8, 9, 10, 11, 12, 13, 14, 15);
        acc0 = wmma16x16x32(a[0][c], bf, acc0);
        acc1 = wmma16x16x32(a[1][c], bf, acc1);
      }
      const float en   = ldsENorm[ct * 16 + lmod];
      const int   kidx = ch * KCHUNK + ct * 16 + lmod;
#pragma unroll
      for (int r = 0; r < 8; ++r) {
        // dist (minus per-row constant ||x||^2) = ||e||^2 - 2 x.e
        float d0 = en - 2.f * acc0[r];
        if (d0 < vmin[0][r]) { vmin[0][r] = d0; vidx[0][r] = kidx; }
        float d1 = en - 2.f * acc1[r];
        if (d1 < vmin[1][r]) { vmin[1][r] = d1; vidx[1][r] = kidx; }
      }
    }
  }

  // ---- cross-lane argmin: 16 lanes (same hh) hold the 16 columns of each row ----
#pragma unroll
  for (int mt = 0; mt < 2; ++mt)
#pragma unroll
    for (int r = 0; r < 8; ++r) {
      float v = vmin[mt][r];
      int   ix = vidx[mt][r];
#pragma unroll
      for (int m = 1; m <= 8; m <<= 1) {
        float ov = __shfl_xor(v, m, 32);
        int   oi = __shfl_xor(ix, m, 32);
        if (ov < v || (ov == v && oi < ix)) { v = ov; ix = oi; }  // first-min ties
      }
      if (lmod == 0)
        ldsIdx[wave * 32 + mt * 16 + hh * 8 + r] = ix;  // C-layout: M = r + 8*hh
    }
  __syncthreads();

  // ---- gather codewords, emit quantized_st + indices, accumulate loss ----
  float lsum = 0.f;
#pragma unroll 4
  for (int i = 0; i < 32; ++i) {
    const int g   = i * 256 + t;   // float4 slot within 256x128 tile (coalesced)
    const int row = g >> 5;
    const int c4  = g & 31;
    const int ix  = ldsIdx[row];
    float4 e  = *(const float4*)(emb + (size_t)ix * D_DIM + c4 * 4);
    float4 xx = *(const float4*)(x + (size_t)(rowBase + row) * D_DIM + c4 * 4);
    float4 q;
    q.x = xx.x + (e.x - xx.x); q.y = xx.y + (e.y - xx.y);  // match ref rounding
    q.z = xx.z + (e.z - xx.z); q.w = xx.w + (e.w - xx.w);
    *(float4*)(q_out + (size_t)(rowBase + row) * D_DIM + c4 * 4) = q;
    float dx = q.x - xx.x, dy = q.y - xx.y, dz = q.z - xx.z, dw = q.w - xx.w;
    lsum += dx * dx + dy * dy + dz * dz + dw * dw;
  }
  idx_out[rowBase + t] = (float)ldsIdx[t];

  // deterministic block loss reduction (fixed-order butterfly + fixed-order sum)
#pragma unroll
  for (int m = 16; m >= 1; m >>= 1) lsum += __shfl_xor(lsum, m, 32);
  if (lane == 0) ldsRed[wave] = lsum;
  __syncthreads();
  if (t == 0) {
    float s = 0.f;
#pragma unroll
    for (int w = 0; w < 8; ++w) s += ldsRed[w];
    part[blockIdx.x] = s;
  }
}

__global__ __launch_bounds__(32) void vq_loss_kernel(
    const float* __restrict__ part, float* __restrict__ loss_out) {
  if (threadIdx.x == 0) {
    float s = 0.f;
    for (int i = 0; i < NBLOCKS; ++i) s += part[i];  // fixed order: deterministic
    loss_out[0] = 0.25f * (s / (float)(N_TOK * D_DIM));
  }
}

extern "C" void kernel_launch(void* const* d_in, const int* in_sizes, int n_in,
                              void* d_out, int out_size, void* d_ws, size_t ws_size,
                              hipStream_t stream) {
  (void)in_sizes; (void)n_in; (void)out_size; (void)ws_size;
  const float* x   = (const float*)d_in[0];   // [N_TOK, 128] f32
  const float* emb = (const float*)d_in[1];   // [1024, 128] f32
  float* q_out   = (float*)d_out;                       // N_TOK*128
  float* idx_out = q_out + (size_t)N_TOK * D_DIM;       // N_TOK (indices as f32)
  float* loss    = idx_out + N_TOK;                     // 1
  float* part    = (float*)d_ws;                        // NBLOCKS partial losses

  vq_main_kernel<<<NBLOCKS, 256, 0, stream>>>(x, emb, q_out, idx_out, part);
  vq_loss_kernel<<<1, 32, 0, stream>>>(part, loss);
}
